// LocalRnn_69827578298736
// MI455X (gfx1250) — compile-verified, matrix operand order checked by reading, not run on hardware
//
#include <hip/hip_runtime.h>
#include <hip/hip_bf16.h>

typedef _Float16 h4  __attribute__((ext_vector_type(4)));
typedef _Float16 h8  __attribute__((ext_vector_type(8)));
typedef _Float16 v16h __attribute__((ext_vector_type(16)));
typedef float    f4  __attribute__((ext_vector_type(4)));
typedef float    v8f __attribute__((ext_vector_type(8)));

constexpr int Dm = 768;    // hidden dim
constexpr int Lm = 4096;   // sequence length
constexpr int Bm = 4;      // batch
constexpr int KC = 192;    // K panel staged in LDS
constexpr int LDP = 200;   // padded LDS row (f16 elems): 400B -> conflict-free b128 reads

// -------- Kernel 0: x[B,D,L] (f32) -> Xt[B*L, D] (f16) tiled transpose --------
__global__ __launch_bounds__(256)
void transpose_to_f16(const float* __restrict__ x, _Float16* __restrict__ Xt) {
    __shared__ _Float16 tile[32][33];
    const int b = blockIdx.z, d0 = blockIdx.y * 32, l0 = blockIdx.x * 32;
    const int tx = threadIdx.x, ty = threadIdx.y;  // (32, 8)
#pragma unroll
    for (int q = 0; q < 4; ++q) {
        int d = d0 + ty + q * 8;
        tile[ty + q * 8][tx] = (_Float16)x[((size_t)b * Dm + d) * Lm + l0 + tx];
    }
    __syncthreads();
#pragma unroll
    for (int q = 0; q < 4; ++q) {
        int t = l0 + ty + q * 8;
        Xt[((size_t)b * Lm + t) * Dm + d0 + tx] = tile[tx][ty + q * 8];
    }
}

// -------- Kernel 0b: weights f32 -> f16 (1.1 MB; stays L2-resident) --------
__global__ __launch_bounds__(256)
void convert_w_f16(const float* __restrict__ W, _Float16* __restrict__ Wh) {
    int i = (blockIdx.x * 256 + threadIdx.x) * 4;
    f4 v = *(const f4*)(W + i);
    *(h4*)(Wh + i) = __builtin_convertvector(v, h4);
}

// -------- WMMA GEMM: C[64x64] per WG, 4 waves, f16 A/B, f32 accumulate --------
// MODE 0: U = A(Xt)*W_ih^T + b_ih + b_hh ; store U (f32) and relu(U) (f16)
// MODE 1: S = relu(U + shift(A=R)*W_hh^T) ; store S (f16)
// MODE 2: E = relu(U + shift(A=S)*W_hh^T) ; store E transposed into [B,D,L] (f32)
template <int MODE>
__global__ __launch_bounds__(128, 1)
void gemm_wmma(const _Float16* __restrict__ A, const _Float16* __restrict__ Wh,
               const float* __restrict__ bi, const float* __restrict__ bh,
               const float* __restrict__ Uin, float* __restrict__ Uout,
               _Float16* __restrict__ Hout, float* __restrict__ Eout) {
    __shared__ _Float16 lds_a[64 * LDP];
    __shared__ _Float16 lds_b[64 * LDP];

    const int tid = threadIdx.x;
    const int w = tid >> 5, lane = tid & 31;
    const int n0 = blockIdx.x * 64;   // output-feature tile
    const int m0 = blockIdx.y * 64;   // (b,t) row tile (64 | 4096 => never straddles b)

    v8f acc[4] = {};

    for (int kk = 0; kk < Dm / KC; ++kk) {
        const int k0 = kk * KC;
        // ---- stage A panel via async global->LDS DMA (ASYNCcnt path) ----
        // 64 rows x 192 f16 = 24 x 16B chunks per row; 12 chunks per thread
#pragma unroll
        for (int it = 0; it < 12; ++it) {
            int idx = it * 128 + tid;
            int r = idx / 24, c = idx % 24;
            int m = m0 + r;
            unsigned lds_addr = (unsigned)(size_t)&lds_a[r * LDP + c * 8];
            bool zero = (MODE != 0) && ((m & (Lm - 1)) == 0);  // shift in zeros at t==0
            if (zero) {
                *(h8*)&lds_a[r * LDP + c * 8] = h8{};
            } else {
                const _Float16* src =
                    A + (size_t)(MODE == 0 ? m : m - 1) * Dm + k0 + c * 8;
                asm volatile("global_load_async_to_lds_b128 %0, %1, off"
                             :: "v"(lds_addr),
                                "v"((unsigned long long)(size_t)src)
                             : "memory");
            }
        }
        // ---- stage B panel (f16 weights) via async DMA: same shape ----
#pragma unroll
        for (int it = 0; it < 12; ++it) {
            int idx = it * 128 + tid;
            int r = idx / 24, c = idx % 24;
            unsigned lds_addr = (unsigned)(size_t)&lds_b[r * LDP + c * 8];
            const _Float16* src = Wh + (size_t)(n0 + r) * Dm + k0 + c * 8;
            asm volatile("global_load_async_to_lds_b128 %0, %1, off"
                         :: "v"(lds_addr),
                            "v"((unsigned long long)(size_t)src)
                         : "memory");
        }
        asm volatile("s_wait_asynccnt 0x0" ::: "memory");
        __syncthreads();

        const int rl = lane & 15;
        const int kh = (lane >> 4) * 8;  // ISA 16-bit A/B lane layout: K-half select
        const int rowa = w * 16 + rl;
#pragma unroll
        for (int ks = 0; ks < KC / 32; ++ks) {
            const _Float16* ap = &lds_a[rowa * LDP + ks * 32];
            h8 alo = *(const h8*)(ap + kh);
            h8 ahi = *(const h8*)(ap + 16 + kh);
            v16h av = __builtin_shufflevector(alo, ahi, 0, 1, 2, 3, 4, 5, 6, 7,
                                              8, 9, 10, 11, 12, 13, 14, 15);
#pragma unroll
            for (int j = 0; j < 4; ++j) {
                const _Float16* bp = &lds_b[(j * 16 + rl) * LDP + ks * 32];
                h8 blo = *(const h8*)(bp + kh);
                h8 bhi = *(const h8*)(bp + 16 + kh);
                v16h bv = __builtin_shufflevector(blo, bhi, 0, 1, 2, 3, 4, 5, 6, 7,
                                                  8, 9, 10, 11, 12, 13, 14, 15);
                acc[j] = __builtin_amdgcn_wmma_f32_16x16x32_f16(
                    false, av, false, bv, (short)0, acc[j], false, false);
            }
        }
        __syncthreads();
    }

    // ---- epilogue: C layout lane(m = half*8 + v, n = lane&15) ----
    const int mb = m0 + w * 16 + (lane >> 4) * 8;
    const int nl = lane & 15;
#pragma unroll
    for (int j = 0; j < 4; ++j) {
        const int n = n0 + j * 16 + nl;
        if (MODE == 0) {
            float bias = bi[n] + bh[n];
#pragma unroll
            for (int v = 0; v < 8; ++v) {
                size_t o = (size_t)(mb + v) * Dm + n;
                float u = acc[j][v] + bias;
                Uout[o] = u;
                Hout[o] = (_Float16)fmaxf(u, 0.f);
            }
        } else if (MODE == 1) {
#pragma unroll
            for (int v = 0; v < 8; ++v) {
                size_t o = (size_t)(mb + v) * Dm + n;
                Hout[o] = (_Float16)fmaxf(Uin[o] + acc[j][v], 0.f);
            }
        } else {
            float ev[8];
#pragma unroll
            for (int v = 0; v < 8; ++v)
                ev[v] = fmaxf(Uin[(size_t)(mb + v) * Dm + n] + acc[j][v], 0.f);
            const int bb = mb >> 12, t0 = mb & (Lm - 1);
            float* dst = Eout + (((size_t)bb * Dm + n) << 12) + t0;  // [B,D,L], t-contig
            *(f4*)dst = f4{ev[0], ev[1], ev[2], ev[3]};
            *(f4*)(dst + 4) = f4{ev[4], ev[5], ev[6], ev[7]};
        }
    }
}

// -------- Kernel 4: out = LayerNorm_L(E + x) * gamma + beta --------
__global__ __launch_bounds__(256)
void residual_layernorm(const float* __restrict__ E, const float* __restrict__ x,
                        const float* __restrict__ gamma, const float* __restrict__ beta,
                        float* __restrict__ out) {
    __shared__ float r1[256], r2[256];
    const size_t base = (size_t)blockIdx.x * Lm;
    const int tid = threadIdx.x;
    float zv[16], s = 0.f, sq = 0.f;
#pragma unroll
    for (int i = 0; i < 16; ++i) {
        int l = tid + i * 256;
        float z = E[base + l] + x[base + l];
        zv[i] = z; s += z; sq += z * z;
    }
    r1[tid] = s; r2[tid] = sq;
    __syncthreads();
    for (int st = 128; st > 0; st >>= 1) {
        if (tid < st) { r1[tid] += r1[tid + st]; r2[tid] += r2[tid + st]; }
        __syncthreads();
    }
    float mean = r1[0] * (1.f / Lm);
    float var = r2[0] * (1.f / Lm) - mean * mean;
    float rstd = rsqrtf(var + 1e-12f);
#pragma unroll
    for (int i = 0; i < 16; ++i) {
        int l = tid + i * 256;
        out[base + l] = (zv[i] - mean) * rstd * gamma[l] + beta[l];
    }
}

extern "C" void kernel_launch(void* const* d_in, const int* in_sizes, int n_in,
                              void* d_out, int out_size, void* d_ws, size_t ws_size,
                              hipStream_t stream) {
    (void)in_sizes; (void)n_in; (void)out_size; (void)ws_size;
    const float* x     = (const float*)d_in[0];
    const float* W_ih  = (const float*)d_in[1];
    const float* W_hh  = (const float*)d_in[2];
    const float* b_ih  = (const float*)d_in[3];
    const float* b_hh  = (const float*)d_in[4];
    const float* gamma = (const float*)d_in[5];
    const float* beta  = (const float*)d_in[6];
    float* out = (float*)d_out;

    const size_t MB = 1024 * 1024;
    char* ws = (char*)d_ws;
    _Float16* Xt  = (_Float16*)ws;               // 24 MB (dead after GEMM1)
    float*    E   = (float*)ws;                  // 48 MB (reuses Xt region)
    float*    U   = (float*)(ws + 48 * MB);      // 48 MB
    _Float16* R   = (_Float16*)(ws + 96 * MB);   // 24 MB (dead after GEMM2)
    _Float16* S   = (_Float16*)(ws + 120 * MB);  // 24 MB
    _Float16* Whi = (_Float16*)(ws + 144 * MB);  // 1.125 MB f16 W_ih
    _Float16* Whh = (_Float16*)(ws + 146 * MB);  // 1.125 MB f16 W_hh

    dim3 gT(Lm / 32, Dm / 32, Bm), bT(32, 8);
    transpose_to_f16<<<gT, bT, 0, stream>>>(x, Xt);
    convert_w_f16<<<(Dm * Dm) / 1024, 256, 0, stream>>>(W_ih, Whi);
    convert_w_f16<<<(Dm * Dm) / 1024, 256, 0, stream>>>(W_hh, Whh);

    dim3 gG(Dm / 64, (Bm * Lm) / 64);
    gemm_wmma<0><<<gG, 128, 0, stream>>>(Xt, Whi, b_ih, b_hh, nullptr, U, R, nullptr);
    gemm_wmma<1><<<gG, 128, 0, stream>>>(R, Whh, nullptr, nullptr, U, nullptr, S, nullptr);
    gemm_wmma<2><<<gG, 128, 0, stream>>>(S, Whh, nullptr, nullptr, U, nullptr, nullptr, E);

    residual_layernorm<<<Bm * Dm, 256, 0, stream>>>(E, x, gamma, beta, out);
}
